// Seq2Seq_attn_80178449482306
// MI455X (gfx1250) — matrix operand order repaired and checked by reading, not compiled, not run on hardware
//
#include <hip/hip_runtime.h>
#include <hip/hip_bf16.h>

#define H     1024
#define E     300
#define VOUT  32000
#define SRC_L 128
#define TGT_L 128
#define G4    (4*H)      // 4096 gate rows
#define K2    (2*H)      // 2048 (concat [h;weighted])
#define NWG   16         // workgroups in persistent LSTM kernels
#define NB    4          // N-tiles per wave in the WMMA GEMM (16x64 strip)

typedef __attribute__((ext_vector_type(16))) __bf16 v16bf;
typedef __attribute__((ext_vector_type(8)))  float  v8f;

// ---------------------------------------------------------------- utilities

__global__ void cvt_f32_to_bf16_kernel(const float* __restrict__ in,
                                       __hip_bfloat16* __restrict__ out, int n) {
  for (int i = blockIdx.x * blockDim.x + threadIdx.x; i < n; i += gridDim.x * blockDim.x)
    out[i] = __float2bfloat16(in[i]);
}

// out[c*rows + r] = in[r*cols + c]
__global__ void transpose_kernel(const float* __restrict__ in, float* __restrict__ out,
                                 int rows, int cols) {
  int n = rows * cols;
  for (int i = blockIdx.x * blockDim.x + threadIdx.x; i < n; i += gridDim.x * blockDim.x) {
    int r = i / cols, c = i - r * cols;
    out[c * rows + r] = in[i];
  }
}

__device__ __forceinline__ float sigmoid_f(float x) { return 1.0f / (1.0f + __expf(-x)); }

// monotonic-epoch grid barrier (all NWG blocks are trivially co-resident)
__device__ __forceinline__ void grid_barrier(unsigned* ctr, unsigned& epoch) {
  __threadfence();
  __syncthreads();
  if (threadIdx.x == 0) {
    __hip_atomic_fetch_add(ctr, 1u, __ATOMIC_RELEASE, __HIP_MEMORY_SCOPE_AGENT);
    unsigned want = (++epoch) * (unsigned)gridDim.x;
    while (__hip_atomic_load(ctr, __ATOMIC_ACQUIRE, __HIP_MEMORY_SCOPE_AGENT) < want)
      __builtin_amdgcn_s_sleep(1);
  }
  __syncthreads();
}

// ------------------------------------------------- persistent sequential LSTM
// 16 WGs x 256 thr = 4096 threads; thread g owns gate row g of W = [Wih|Whh].
// Weights pre-transposed: WihT[j*4096 + r], WhhT[j*4096 + r]  (coalesced in r).
__global__ void lstm_seq_kernel(const float* __restrict__ WihT,
                                const float* __restrict__ WhhT,
                                const float* __restrict__ bias,
                                const float* __restrict__ embed,   // [vocab, E]
                                const int*   __restrict__ tokens,  // [steps]
                                float* __restrict__ h_buf,         // [H] in/out
                                float* __restrict__ c_buf,         // [H] in/out
                                float* __restrict__ outs,          // [steps, H]
                                float* __restrict__ gates,         // [4H] scratch
                                unsigned* __restrict__ bar, int steps) {
  __shared__ float sx[E];
  __shared__ float sh[H];
  const int tid  = threadIdx.x;
  const int gtid = blockIdx.x * blockDim.x + tid;  // gate row, 0..4095
  const float b  = bias[gtid];
  unsigned epoch = 0;

  for (int t = 0; t < steps; ++t) {
    // stage x_t and h into LDS
    const int tok = tokens[t];
    for (int j = tid; j < E; j += blockDim.x) sx[j] = embed[tok * E + j];
    for (int j = tid; j < H; j += blockDim.x) sh[j] = h_buf[j];
    __syncthreads();

    float acc = b;
    #pragma unroll 4
    for (int j = 0; j < E; ++j) acc = fmaf(WihT[j * G4 + gtid], sx[j], acc);
    #pragma unroll 4
    for (int j = 0; j < H; ++j) acc = fmaf(WhhT[j * G4 + gtid], sh[j], acc);
    gates[gtid] = acc;

    grid_barrier(bar, epoch);

    if (gtid < H) {
      float gi = sigmoid_f(gates[gtid]);
      float gf = sigmoid_f(gates[gtid + H]);
      float gg = tanhf    (gates[gtid + 2 * H]);
      float go = sigmoid_f(gates[gtid + 3 * H]);
      float c  = gf * c_buf[gtid] + gi * gg;
      float h  = go * tanhf(c);
      c_buf[gtid] = c;
      h_buf[gtid] = h;
      outs[t * H + gtid] = h;
    }
    grid_barrier(bar, epoch);
  }
}

// ------------------------------------------------- batched attention (small)
// V[t,i] = sum_k mult_W[i,k] * Hdec[t,k]   via multWT[k*H+i] (coalesced)
__global__ void vmat_kernel(const float* __restrict__ multWT,
                            const float* __restrict__ Hdec,
                            float* __restrict__ Vmat) {
  __shared__ float shh[H];
  const int t = blockIdx.y;
  const int i = blockIdx.x * blockDim.x + threadIdx.x;
  for (int j = threadIdx.x; j < H; j += blockDim.x) shh[j] = Hdec[t * H + j];
  __syncthreads();
  float acc = 0.f;
  #pragma unroll 4
  for (int k = 0; k < H; ++k) acc = fmaf(multWT[k * H + i], shh[k], acc);
  Vmat[t * H + i] = acc;
}

// per target step t: scores over SRC, softmax, weighted sum; emit bf16 X row.
__global__ void attn_kernel(const float* __restrict__ Vmat,
                            const float* __restrict__ EncOuts,
                            const float* __restrict__ Hdec,
                            __hip_bfloat16* __restrict__ Xb) {   // [TGT, 2H]
  __shared__ float sv[H];
  __shared__ float sc[SRC_L];
  __shared__ float red[256];
  const int t = blockIdx.x, tid = threadIdx.x;
  for (int j = tid; j < H; j += blockDim.x) sv[j] = Vmat[t * H + j];
  __syncthreads();

  if (tid < SRC_L) {
    float acc = 0.f;
    #pragma unroll 4
    for (int k = 0; k < H; ++k) acc = fmaf(EncOuts[tid * H + k], sv[k], acc);
    sc[tid] = acc;
  }
  __syncthreads();

  // max reduce
  red[tid] = (tid < SRC_L) ? sc[tid] : -3.4e38f;
  __syncthreads();
  for (int s = 128; s > 0; s >>= 1) {
    if (tid < s) red[tid] = fmaxf(red[tid], red[tid + s]);
    __syncthreads();
  }
  const float m = red[0];
  __syncthreads();
  if (tid < SRC_L) sc[tid] = __expf(sc[tid] - m);
  __syncthreads();
  red[tid] = (tid < SRC_L) ? sc[tid] : 0.f;
  __syncthreads();
  for (int s = 128; s > 0; s >>= 1) {
    if (tid < s) red[tid] += red[tid + s];
    __syncthreads();
  }
  const float inv_sum = 1.0f / red[0];
  __syncthreads();

  for (int u = tid; u < H; u += blockDim.x) {
    float w = 0.f;
    #pragma unroll 4
    for (int s = 0; s < SRC_L; ++s) w = fmaf(sc[s], EncOuts[s * H + u], w);
    w *= inv_sum;
    Xb[t * K2 + u]     = __float2bfloat16(Hdec[t * H + u]);
    Xb[t * K2 + H + u] = __float2bfloat16(w);
  }
}

// ------------------------------------------------- Phase D: WMMA bf16 GEMM
// logits[128,32000] = X[128,2048] * time_W^T + b.
// Each wave owns a 16x64 output strip: one shared A fragment, NB=4 independent
// accumulator chains -> 4 WMMAs per k-step, 2.5 b128 loads per WMMA, and the
// accumulator RAW hazard of one chain is hidden by the other three.
__global__ void logits_wmma_kernel(const __hip_bfloat16* __restrict__ X,
                                   const __hip_bfloat16* __restrict__ Wb, // [VOUT,K2]
                                   const float* __restrict__ time_b,
                                   float* __restrict__ out) {
  const int n0   = blockIdx.x * (16 * NB);
  const int m0   = blockIdx.y * 16;
  const int lane = threadIdx.x;          // 0..31 (wave32)
  const int lo16 = (lane < 16);
  const int lm   = lane & 15;

  const float4* Xf = reinterpret_cast<const float4*>(X);   // 8 bf16 per float4
  const float4* Wf = reinterpret_cast<const float4*>(Wb);

  union Frag { v16bf v; float4 f[2]; };
  v8f acc[NB];
  #pragma unroll
  for (int i = 0; i < NB; ++i) acc[i] = (v8f){};

  // per-lane base element offsets (bf16 elements)
  const int a_row  = (m0 + lm) * K2 + (lo16 ? 0 : 8);      // A: two 8-elem runs
  const int b_base = (n0 + lm) * K2 + (lo16 ? 0 : 8);      // B: two 8-elem runs

  #pragma unroll 2
  for (int kc = 0; kc < K2; kc += 32) {
    // A: 16x32 bf16, row m0+lm; K runs {0..7,16..23} (lanes<16) / {8..15,24..31}
    Frag a;
    a.f[0] = Xf[(a_row + kc) >> 3];
    a.f[1] = Xf[(a_row + kc + 16) >> 3];

    // B: 32x16 bf16 per tile; B[k][n] = time_W[n0+16*i+n][kc+k]
    Frag bm[NB];
    #pragma unroll
    for (int i = 0; i < NB; ++i) {
      const int e0 = b_base + i * 16 * K2 + kc;
      bm[i].f[0] = Wf[e0 >> 3];
      bm[i].f[1] = Wf[(e0 + 16) >> 3];
    }
    __builtin_prefetch(&Wf[(b_base + kc + 64) >> 3], 0, 1);  // global_prefetch_b8

    #pragma unroll
    for (int i = 0; i < NB; ++i)
      acc[i] = __builtin_amdgcn_wmma_f32_16x16x32_bf16(
          /*neg_a=*/false, a.v, /*neg_b=*/false, bm[i].v,
          /*c_mod=*/(short)0, acc[i], /*reuse_a=*/false, /*reuse_b=*/false);
  }

  // D layout: VGPR r, lanes<16 -> row m0+r, lanes>=16 -> row m0+8+r; col = lm
  const int rbase = m0 + (lo16 ? 0 : 8);
  #pragma unroll
  for (int i = 0; i < NB; ++i) {
    const int col  = n0 + i * 16 + lm;
    const float bv = time_b[col];
    #pragma unroll
    for (int r = 0; r < 8; ++r)
      out[(rbase + r) * VOUT + col] = acc[i][r] + bv;
  }
}

// ------------------------------------------------- in-place row log_softmax
__global__ void log_softmax_kernel(float* __restrict__ out) {
  __shared__ float red[256];
  const int t = blockIdx.x, tid = threadIdx.x;
  float* row = out + (size_t)t * VOUT;

  float m = -3.4e38f;
  for (int j = tid; j < VOUT; j += blockDim.x) m = fmaxf(m, row[j]);
  red[tid] = m; __syncthreads();
  for (int s = 128; s > 0; s >>= 1) { if (tid < s) red[tid] = fmaxf(red[tid], red[tid + s]); __syncthreads(); }
  m = red[0]; __syncthreads();

  float sum = 0.f;
  for (int j = tid; j < VOUT; j += blockDim.x) sum += __expf(row[j] - m);
  red[tid] = sum; __syncthreads();
  for (int s = 128; s > 0; s >>= 1) { if (tid < s) red[tid] += red[tid + s]; __syncthreads(); }
  const float lse = m + __logf(red[0]);
  __syncthreads();

  for (int j = tid; j < VOUT; j += blockDim.x) row[j] = row[j] - lse;
}

// ---------------------------------------------------------------- launcher
extern "C" void kernel_launch(void* const* d_in, const int* in_sizes, int n_in,
                              void* d_out, int out_size, void* d_ws, size_t ws_size,
                              hipStream_t stream) {
  const int*   src_tok  = (const int*)  d_in[0];
  const int*   tgt_tok  = (const int*)  d_in[1];
  const float* enc_emb  = (const float*)d_in[2];
  const float* dec_emb  = (const float*)d_in[3];
  const float* enc_Wih  = (const float*)d_in[4];
  const float* enc_Whh  = (const float*)d_in[5];
  const float* enc_b    = (const float*)d_in[6];
  const float* dec_Wih  = (const float*)d_in[7];
  const float* dec_Whh  = (const float*)d_in[8];
  const float* dec_b    = (const float*)d_in[9];
  const float* mult_W   = (const float*)d_in[10];
  const float* time_W   = (const float*)d_in[11];
  const float* time_b   = (const float*)d_in[12];
  float* out = (float*)d_out;

  // bump allocator over workspace (256B aligned)
  char* p = (char*)d_ws;
  auto alloc = [&](size_t bytes) -> char* {
    char* r = p; p += (bytes + 255) & ~(size_t)255; return r;
  };
  __hip_bfloat16* Wb      = (__hip_bfloat16*)alloc((size_t)VOUT * K2 * 2); // 131 MB, L2-resident
  float* encWihT = (float*)alloc((size_t)G4 * E * 4);
  float* encWhhT = (float*)alloc((size_t)G4 * H * 4);
  float* decWihT = (float*)alloc((size_t)G4 * E * 4);
  float* decWhhT = (float*)alloc((size_t)G4 * H * 4);
  float* multWT  = (float*)alloc((size_t)H * H * 4);
  float* EncOuts = (float*)alloc((size_t)SRC_L * H * 4);
  float* Hdec    = (float*)alloc((size_t)TGT_L * H * 4);
  float* Vmat    = (float*)alloc((size_t)TGT_L * H * 4);
  __hip_bfloat16* Xb = (__hip_bfloat16*)alloc((size_t)TGT_L * K2 * 2);
  float* h_buf   = (float*)alloc(H * 4);
  float* c_buf   = (float*)alloc(H * 4);
  float* gates   = (float*)alloc(G4 * 4);
  unsigned* bars = (unsigned*)alloc(256);          // two barrier counters

  // per-launch state init (graph-capturable)
  hipMemsetAsync(h_buf, 0, 2 * H * 4, stream);     // h_buf + c_buf contiguous
  hipMemsetAsync(bars, 0, 256, stream);

  // one-time (per launch) weight prep
  cvt_f32_to_bf16_kernel<<<4096, 256, 0, stream>>>(time_W, Wb, VOUT * K2);
  transpose_kernel<<<2048, 256, 0, stream>>>(enc_Wih, encWihT, G4, E);
  transpose_kernel<<<2048, 256, 0, stream>>>(enc_Whh, encWhhT, G4, H);
  transpose_kernel<<<2048, 256, 0, stream>>>(dec_Wih, decWihT, G4, E);
  transpose_kernel<<<2048, 256, 0, stream>>>(dec_Whh, decWhhT, G4, H);
  transpose_kernel<<<1024, 256, 0, stream>>>(mult_W,  multWT,  H,  H);

  // Phase A: encoder recurrence (leaves final h,c in h_buf/c_buf)
  lstm_seq_kernel<<<NWG, 256, 0, stream>>>(encWihT, encWhhT, enc_b, enc_emb, src_tok,
                                           h_buf, c_buf, EncOuts, gates, bars, SRC_L);
  // Phase B: decoder recurrence (seeded by encoder final state)
  lstm_seq_kernel<<<NWG, 256, 0, stream>>>(decWihT, decWhhT, dec_b, dec_emb, tgt_tok,
                                           h_buf, c_buf, Hdec, gates, bars + 64, TGT_L);
  // Phase C: batched attention -> Xb = bf16([Hdec | weighted])
  vmat_kernel<<<dim3(H / 256, TGT_L), 256, 0, stream>>>(multWT, Hdec, Vmat);
  attn_kernel<<<TGT_L, 256, 0, stream>>>(Vmat, EncOuts, Hdec, Xb);
  // Phase D: 128x32000x2048 bf16 WMMA GEMM (+bias) into d_out, 16x64 per wave
  logits_wmma_kernel<<<dim3(VOUT / (16 * NB), TGT_L / 16), 32, 0, stream>>>(Xb, Wb, time_b, out);
  // Phase E: in-place row-wise log_softmax
  log_softmax_kernel<<<TGT_L, 256, 0, stream>>>(out);

  (void)in_sizes; (void)n_in; (void)out_size; (void)ws_size;
}
